// OldCoarseToFineCursorDecoder_45835890983342
// MI455X (gfx1250) — matrix operand order repaired and checked by reading, not compiled, not run on hardware
//
#include <hip/hip_runtime.h>
#include <hip/hip_bf16.h>
#include <math.h>

// ---------------------------------------------------------------------------
// CoarseToFineCursorDecoder for MI455X (gfx1250, wave32, WMMA)
//   - all GEMMs: v_wmma_f32_16x16x32_f16 (f16 operands, f32 accum)
//   - weights repacked once per launch into WMMA B-fragment order
//   - final 537MB scatter: non-temporal, fully-coalesced fp32 streaming store
//     (keeps coarse/finval + GEMM operands resident in the 192MB L2)
// ---------------------------------------------------------------------------

#define BB   2048
#define CC   512
#define NNC  256     // coarse grid total
#define FFN  256     // fine total
#define KK   4
#define OUTW 65537   // 1 + N*F
#define LOG_F 5.545177444479562f

typedef __attribute__((ext_vector_type(16))) _Float16 v16h;
typedef __attribute__((ext_vector_type(8)))  _Float16 v8h;
typedef __attribute__((ext_vector_type(8)))  float    v8f;

__device__ __forceinline__ float gelu_f(float x) {
    const float k = 0.7978845608028654f;
    float x3 = x * x * x;
    return 0.5f * x * (1.0f + tanhf(k * (x + 0.044715f * x3)));
}

__device__ __forceinline__ float block_reduce_sum(float v, float* sm) {
    int t = threadIdx.x;
    sm[t] = v; __syncthreads();
    for (int s = 128; s > 0; s >>= 1) {
        if (t < s) sm[t] += sm[t + s];
        __syncthreads();
    }
    float r = sm[0]; __syncthreads();
    return r;
}

// --- LayerNorm of x -> xn (f16), fused no_op column into out[:,0] -----------
__global__ void ln_input_kernel(const float* __restrict__ x,
                                const float* __restrict__ g,
                                const float* __restrict__ b,
                                const float* __restrict__ nop_w,
                                const float* __restrict__ nop_b,
                                _Float16* __restrict__ xn_h,
                                float* __restrict__ out) {
    __shared__ float sm[256];
    int row = blockIdx.x, t = threadIdx.x;
    const float* xr = x + (size_t)row * CC;
    float a0 = __builtin_nontemporal_load(xr + t);        // x read exactly once
    float a1 = __builtin_nontemporal_load(xr + t + 256);
    float mean = block_reduce_sum(a0 + a1, sm) * (1.0f / CC);
    float d0 = a0 - mean, d1 = a1 - mean;
    float var = block_reduce_sum(d0 * d0 + d1 * d1, sm) * (1.0f / CC);
    float inv = rsqrtf(var + 1e-5f);
    float y0 = d0 * inv * g[t] + b[t];
    float y1 = d1 * inv * g[t + 256] + b[t + 256];
    xn_h[(size_t)row * CC + t]       = (_Float16)y0;
    xn_h[(size_t)row * CC + t + 256] = (_Float16)y1;
    float dot = block_reduce_sum(y0 * nop_w[t] + y1 * nop_w[t + 256], sm);
    if (t == 0) __builtin_nontemporal_store(dot + nop_b[0], out + (size_t)row * OUTW);
}

// --- repack fp32 weight (Ktot x Ntot, row-major) into WMMA B-fragment order -
// layout: [ntile][ktile][lane 0..31][16 halfs]; lane holds col n = nt*16+(l&15),
// K = kt*32 + (e>>3)*16 + (l>>4)*8 + (e&7)  for e = 0..15  (matches 16-bit B 32x16 layout)
__global__ void repack_w_kernel(const float* __restrict__ W,
                                _Float16* __restrict__ Bp,
                                int Ktot, int Ntot) {
    int tid = blockIdx.x * 256 + threadIdx.x;
    int e = tid & 15;
    int lane = (tid >> 4) & 31;
    int ktnt = tid >> 9;
    int nKt = Ktot >> 5;
    int kt = ktnt % nKt;
    int nt = ktnt / nKt;
    int k = (kt << 5) + ((e >> 3) << 4) + ((lane >> 4) << 3) + (e & 7);
    int n = (nt << 4) + (lane & 15);
    Bp[tid] = (_Float16)W[(size_t)k * Ntot + n];
}

// --- wave-per-tile WMMA GEMM: out(M x Ntot) = A(M x Ktot) @ Bp + bias -------
// 8 waves / block; gridDim.x = Ntot/16; gridDim.y = M/128.
template <bool GELU>
__global__ void gemm_wmma_kernel(const _Float16* __restrict__ A,
                                 const _Float16* __restrict__ Bp,
                                 const float* __restrict__ bias,
                                 void* __restrict__ outp,
                                 int Ktot, int Ntot) {
    int lane  = threadIdx.x & 31;
    int wave  = threadIdx.x >> 5;
    int ntile = blockIdx.x;
    int mtile = blockIdx.y * 8 + wave;
    int nKt   = Ktot >> 5;

    // A fragment addressing (16-bit A 16x32): lane<16 -> K {0..7,16..23},
    // lane>=16 -> K {8..15,24..31}; two contiguous 16B loads per step.
    int row  = lane & 15;
    int koff = (lane >> 4) << 3;
    const _Float16* Arow = A + (size_t)(mtile * 16 + row) * Ktot + koff;
    const _Float16* Bl   = Bp + ((size_t)(ntile * nKt) * 32 + lane) * 16;

    v8f acc = {};
    for (int kt = 0; kt < nKt; ++kt) {
        union { v16h v; v8h h[2]; } a;
        a.h[0] = *(const v8h*)(Arow);
        a.h[1] = *(const v8h*)(Arow + 16);
        v16h bf = *(const v16h*)(Bl);
        acc = __builtin_amdgcn_wmma_f32_16x16x32_f16(
            false, a.v, false, bf, (short)0, acc, false, false);
        Arow += 32;
        Bl   += 32 * 16;
    }

    // C/D layout: VGPR r -> M = r + 8*(lane>>4), N = lane&15
    int col = lane & 15;
    int m0  = mtile * 16 + ((lane >> 4) << 3);
    int n   = ntile * 16 + col;
    float bv = bias[n];
    if (GELU) {
        _Float16* out = (_Float16*)outp;
#pragma unroll
        for (int r = 0; r < 8; ++r)
            out[(size_t)(m0 + r) * Ntot + n] = (_Float16)gelu_f(acc[r] + bv);
    } else {
        float* out = (float*)outp;
#pragma unroll
        for (int r = 0; r < 8; ++r)
            out[(size_t)(m0 + r) * Ntot + n] = acc[r] + bv;
    }
}

// --- top-4 of coarse row: one wave32 per row --------------------------------
__global__ void topk_kernel(const float* __restrict__ coarse,
                            int* __restrict__ idx,
                            float* __restrict__ catidx) {
    int wave = threadIdx.x >> 5, lane = threadIdx.x & 31;
    int bidx = blockIdx.x * 8 + wave;
    const float* cr = coarse + (size_t)bidx * NNC;
    float v[8]; bool taken[8];
#pragma unroll
    for (int t = 0; t < 8; ++t) { v[t] = cr[lane + 32 * t]; taken[t] = false; }
    for (int sel = 0; sel < KK; ++sel) {
        float m = -3.4e38f; int mt = -1;
#pragma unroll
        for (int t = 0; t < 8; ++t)
            if (!taken[t] && v[t] > m) { m = v[t]; mt = t; }
        int mi = (mt < 0) ? 0x7fffffff : (lane + 32 * mt);
        for (int off = 16; off > 0; off >>= 1) {
            float om = __shfl_xor(m, off, 32);
            int   oi = __shfl_xor(mi, off, 32);
            if (om > m || (om == m && oi < mi)) { m = om; mi = oi; }
        }
        if (lane == 0) { idx[bidx * KK + sel] = mi; catidx[bidx * KK + sel] = m; }
        if ((mi & 31) == lane && (mi >> 5) < 8) taken[mi >> 5] = true;
    }
}

// --- gather emb[idx], LN, concat with xn -> fin (f16, 8192 x 1024) ----------
__global__ void build_fin_kernel(const _Float16* __restrict__ xn_h,
                                 const float* __restrict__ emb,
                                 const float* __restrict__ eg,
                                 const float* __restrict__ eb,
                                 const int* __restrict__ idx,
                                 _Float16* __restrict__ fin) {
    __shared__ float sm[256];
    int rk = blockIdx.x;        // b*4 + k
    int b  = rk >> 2;
    int t  = threadIdx.x;
    int id = idx[rk];
    const float* er = emb + (size_t)id * CC;
    float a0 = er[t], a1 = er[t + 256];
    float mean = block_reduce_sum(a0 + a1, sm) * (1.0f / CC);
    float d0 = a0 - mean, d1 = a1 - mean;
    float var = block_reduce_sum(d0 * d0 + d1 * d1, sm) * (1.0f / CC);
    float inv = rsqrtf(var + 1e-5f);
    _Float16* fr = fin + (size_t)rk * (2 * CC);
    fr[t]        = xn_h[(size_t)b * CC + t];
    fr[t + 256]  = xn_h[(size_t)b * CC + t + 256];
    fr[CC + t]       = (_Float16)(d0 * inv * eg[t] + eb[t]);
    fr[CC + t + 256] = (_Float16)(d1 * inv * eg[t + 256] + eb[t + 256]);
}

// --- in-place: fine[rk,f] = coarse[b,idx] + log_softmax(fine[rk,:])[f] ------
__global__ void softmax_fixup_kernel(float* __restrict__ fine,
                                     const float* __restrict__ catidx) {
    __shared__ float sm[256];
    int rk = blockIdx.x, t = threadIdx.x;
    float* fr = fine + (size_t)rk * FFN;
    float v = fr[t];
    sm[t] = v; __syncthreads();
    for (int s = 128; s > 0; s >>= 1) {
        if (t < s) sm[t] = fmaxf(sm[t], sm[t + s]);
        __syncthreads();
    }
    float mx = sm[0]; __syncthreads();
    float se = block_reduce_sum(__expf(v - mx), sm);
    fr[t] = catidx[rk] + (v - mx - logf(se));
}

// --- the HBM-bound writer: 2048 x 65536 fp32, coalesced + non-temporal ------
// out index o -> (c0,f0,c1,f1); n = c0*16+c1, f = f0*16+f1
__global__ void scatter_out_kernel(const float* __restrict__ coarse,
                                   const int* __restrict__ idx,
                                   const float* __restrict__ finval,
                                   float* __restrict__ out) {
    int b = blockIdx.y;
    int o = blockIdx.x * 256 + threadIdx.x;
    int f1 = o & 15, c1 = (o >> 4) & 15, f0 = (o >> 8) & 15, c0 = o >> 12;
    int n = (c0 << 4) | c1;
    int f = (f0 << 4) | f1;
    const int4 id = *(const int4*)(idx + b * KK);
    float v = coarse[(size_t)b * NNC + n] - LOG_F;
    const float* fv = finval + (size_t)b * KK * FFN;
    if      (n == id.x) v = fv[0 * FFN + f];
    else if (n == id.y) v = fv[1 * FFN + f];
    else if (n == id.z) v = fv[2 * FFN + f];
    else if (n == id.w) v = fv[3 * FFN + f];
    // write-once stream: NT store keeps L2 for the reused coarse/finval rows
    __builtin_nontemporal_store(v, out + (size_t)b * OUTW + 1 + o);
}

// ---------------------------------------------------------------------------
extern "C" void kernel_launch(void* const* d_in, const int* in_sizes, int n_in,
                              void* d_out, int out_size, void* d_ws, size_t ws_size,
                              hipStream_t stream) {
    (void)in_sizes; (void)n_in; (void)out_size; (void)ws_size;
    const float* x     = (const float*)d_in[0];
    const float* in_g  = (const float*)d_in[1];
    const float* in_b  = (const float*)d_in[2];
    const float* nop_w = (const float*)d_in[3];
    const float* nop_b = (const float*)d_in[4];
    const float* c_w1  = (const float*)d_in[5];
    const float* c_b1  = (const float*)d_in[6];
    const float* c_w2  = (const float*)d_in[7];
    const float* c_b2  = (const float*)d_in[8];
    const float* c_w3  = (const float*)d_in[9];
    const float* c_b3  = (const float*)d_in[10];
    const float* emb   = (const float*)d_in[11];
    const float* emb_g = (const float*)d_in[12];
    const float* emb_b = (const float*)d_in[13];
    const float* f_w1  = (const float*)d_in[14];
    const float* f_b1  = (const float*)d_in[15];
    const float* f_w2  = (const float*)d_in[16];
    const float* f_b2  = (const float*)d_in[17];
    const float* f_w3  = (const float*)d_in[18];
    const float* f_b3  = (const float*)d_in[19];
    float* out = (float*)d_out;

    char* ws = (char*)d_ws;
    size_t off = 0;
    auto alloc = [&](size_t bytes) -> char* {
        char* p = ws + off;
        off = (off + bytes + 255) & ~(size_t)255;
        return p;
    };
    _Float16* xn_h  = (_Float16*)alloc((size_t)BB * CC * 2);
    _Float16* cw1p  = (_Float16*)alloc((size_t)CC * CC * 2);
    _Float16* cw2p  = (_Float16*)alloc((size_t)CC * CC * 2);
    _Float16* cw3p  = (_Float16*)alloc((size_t)CC * NNC * 2);
    _Float16* fw1p  = (_Float16*)alloc((size_t)2 * CC * CC * 2);
    _Float16* fw2p  = (_Float16*)alloc((size_t)CC * CC * 2);
    _Float16* fw3p  = (_Float16*)alloc((size_t)CC * FFN * 2);
    _Float16* h1    = (_Float16*)alloc((size_t)BB * CC * 2);
    _Float16* h2    = (_Float16*)alloc((size_t)BB * CC * 2);
    float*    coarse= (float*)   alloc((size_t)BB * NNC * 4);
    int*      idx   = (int*)     alloc((size_t)BB * KK * 4);
    float*    catidx= (float*)   alloc((size_t)BB * KK * 4);
    _Float16* fin   = (_Float16*)alloc((size_t)BB * KK * 2 * CC * 2);
    _Float16* fh1   = (_Float16*)alloc((size_t)BB * KK * CC * 2);
    _Float16* fh2   = (_Float16*)alloc((size_t)BB * KK * CC * 2);
    float*    fine  = (float*)   alloc((size_t)BB * KK * FFN * 4);

    // 1) repack all six weight matrices into WMMA fragment order
    repack_w_kernel<<<(CC * CC) / 256, 256, 0, stream>>>(c_w1, cw1p, CC, CC);
    repack_w_kernel<<<(CC * CC) / 256, 256, 0, stream>>>(c_w2, cw2p, CC, CC);
    repack_w_kernel<<<(CC * NNC) / 256, 256, 0, stream>>>(c_w3, cw3p, CC, NNC);
    repack_w_kernel<<<(2 * CC * CC) / 256, 256, 0, stream>>>(f_w1, fw1p, 2 * CC, CC);
    repack_w_kernel<<<(CC * CC) / 256, 256, 0, stream>>>(f_w2, fw2p, CC, CC);
    repack_w_kernel<<<(CC * FFN) / 256, 256, 0, stream>>>(f_w3, fw3p, CC, FFN);

    // 2) input LN (+ no_op column fused into out[:,0])
    ln_input_kernel<<<BB, 256, 0, stream>>>(x, in_g, in_b, nop_w, nop_b, xn_h, out);

    // 3) coarse MLP: 2048x512 @ 512x512 (gelu) @ 512x512 (gelu) @ 512x256
    gemm_wmma_kernel<true ><<<dim3(CC / 16, BB / 128), 256, 0, stream>>>(xn_h, cw1p, c_b1, h1, CC, CC);
    gemm_wmma_kernel<true ><<<dim3(CC / 16, BB / 128), 256, 0, stream>>>(h1,  cw2p, c_b2, h2, CC, CC);
    gemm_wmma_kernel<false><<<dim3(NNC / 16, BB / 128), 256, 0, stream>>>(h2,  cw3p, c_b3, coarse, CC, NNC);

    // 4) top-4 per row
    topk_kernel<<<BB / 8, 256, 0, stream>>>(coarse, idx, catidx);

    // 5) gather + LN + concat -> fin
    build_fin_kernel<<<BB * KK, 256, 0, stream>>>(xn_h, emb, emb_g, emb_b, idx, fin);

    // 6) fine MLP: 8192x1024 @ 1024x512 (gelu) @ 512x512 (gelu) @ 512x256
    gemm_wmma_kernel<true ><<<dim3(CC / 16, (BB * KK) / 128), 256, 0, stream>>>(fin, fw1p, f_b1, fh1, 2 * CC, CC);
    gemm_wmma_kernel<true ><<<dim3(CC / 16, (BB * KK) / 128), 256, 0, stream>>>(fh1, fw2p, f_b2, fh2, CC, CC);
    gemm_wmma_kernel<false><<<dim3(FFN / 16, (BB * KK) / 128), 256, 0, stream>>>(fh2, fw3p, f_b3, fine, CC, FFN);

    // 7) log-softmax fixup (in place), adds coarse[b, idx]
    softmax_fixup_kernel<<<BB * KK, 256, 0, stream>>>(fine, catidx);

    // 8) the 537MB non-temporal coalesced stream-out with (c0,f0,c1,f1) permute
    scatter_out_kernel<<<dim3((NNC * FFN) / 256, BB), 256, 0, stream>>>(coarse, idx, fine, out);
}
